// LoRALinear_20409684591173
// MI455X (gfx1250) — matrix operand order repaired and checked by reading, not compiled, not run on hardware
//
#include <hip/hip_runtime.h>

// ---------------------------------------------------------------------------
// DoRA forward, algebraically collapsed:
//   out[m,o] = (mag[o]/||W_eff[o,:]||) * (x[m,:] . W_eff[o,:])
//   W_eff    = W + SCALING * (B @ A)
// => prep: Wس = diag(mag/rownorm(W_eff)) * W_eff, split into bf16 hi/lo
// => main: out = x @ Wس^T via V_WMMA_F32_16X16X32_BF16 with hi/lo (3xTF32)
// ---------------------------------------------------------------------------

typedef __attribute__((ext_vector_type(16))) __bf16 v16bf;
typedef __attribute__((ext_vector_type(8)))  __bf16 bf16x8;
typedef __attribute__((ext_vector_type(4)))  __bf16 bf16x4;
typedef __attribute__((ext_vector_type(8)))  float  v8f;
typedef __attribute__((ext_vector_type(4)))  float  f32x4;

#define DORA_SCALING 2.0f   // alpha/rank = 32/16
#define LDS_PITCH    40     // 32 bf16 + pad; 80B row stride, 16B-aligned, conflict-free

// ---------------------------------------------------------------------------
// Prep: one block per output row o.
//   weff = W[o,:] + 2 * sum_r b[o,r]*A[r,:]
//   s    = mag[o] / sqrt(sum weff^2)
//   whi/wlo = bf16 split of s*weff
// ---------------------------------------------------------------------------
__global__ __launch_bounds__(256) void dora_prep_kernel(
    const float* __restrict__ W, const float* __restrict__ A,
    const float* __restrict__ Bw, const float* __restrict__ mag,
    __bf16* __restrict__ whi, __bf16* __restrict__ wlo,
    int D_IN, int R)
{
  const int o   = blockIdx.x;
  const int tid = threadIdx.x;

  __shared__ float red[256];
  __shared__ float brow[64];
  if (tid < R) brow[tid] = Bw[(size_t)o * R + tid];
  __syncthreads();

  float weff[4];
  float sumsq = 0.f;
#pragma unroll
  for (int j = 0; j < 4; ++j) {
    int d = tid + j * 256;
    float w = 0.f;
    if (d < D_IN) {
      float lora = 0.f;
      for (int r = 0; r < R; ++r)
        lora += brow[r] * A[(size_t)r * D_IN + d];
      w = W[(size_t)o * D_IN + d] + DORA_SCALING * lora;
    }
    weff[j] = w;
    sumsq += w * w;
  }

  red[tid] = sumsq;
  __syncthreads();
  for (int s = 128; s > 0; s >>= 1) {
    if (tid < s) red[tid] += red[tid + s];
    __syncthreads();
  }
  const float scale = mag[o] / sqrtf(red[0]);

#pragma unroll
  for (int j = 0; j < 4; ++j) {
    int d = tid + j * 256;
    if (d < D_IN) {
      float ws = weff[j] * scale;
      __bf16 h  = (__bf16)ws;
      __bf16 l  = (__bf16)(ws - (float)h);
      whi[(size_t)o * D_IN + d] = h;
      wlo[(size_t)o * D_IN + d] = l;
    }
  }
}

// ---------------------------------------------------------------------------
// Main GEMM: out[M,N] = x[M,K] @ Wس^T, Wس stored row-major [N,K] as bf16 hi/lo.
// Block tile 128x128, 8 waves (wave32), each wave 64x32 = 4x2 WMMA tiles.
// ---------------------------------------------------------------------------
__global__ __launch_bounds__(256) void dora_gemm_kernel(
    const float* __restrict__ x,
    const __bf16* __restrict__ whi, const __bf16* __restrict__ wlo,
    float* __restrict__ out, int K, int N)
{
  __shared__ __bf16 sh_hi[128 * LDS_PITCH];
  __shared__ __bf16 sh_lo[128 * LDS_PITCH];

  const int tid     = threadIdx.x;
  const int lane    = tid & 31;
  const int laneLow = lane & 15;   // row (A) / col (B) within 16x16 fragment
  const int laneHi  = lane >> 4;   // which K-half of the fragment
  const int wid     = tid >> 5;
  const int wave_m  = wid & 1;     // 0..1  -> 64-row band
  const int wave_n  = wid >> 1;    // 0..3  -> 32-col band

  const int mBase = blockIdx.y * 128;
  const int nBase = blockIdx.x * 128;

  // staging decomposition: 256 threads x (8 float4 per 32-wide row)
  const int f4    = tid & 7;       // float4 index within row
  const int rBase = tid >> 3;      // 0..31, +32 per pass

  const v8f vzero = {0.f, 0.f, 0.f, 0.f, 0.f, 0.f, 0.f, 0.f};
  v8f acc[4][2];
#pragma unroll
  for (int mt = 0; mt < 4; ++mt)
#pragma unroll
    for (int nt = 0; nt < 2; ++nt)
      acc[mt][nt] = vzero;

  for (int k0 = 0; k0 < K; k0 += 32) {
    // ---- stage x tile (128 x 32 fp32) -> LDS bf16 hi/lo --------------------
#pragma unroll
    for (int p = 0; p < 4; ++p) {
      const int row = rBase + p * 32;
      const float* gp = x + (size_t)(mBase + row) * K + k0 + f4 * 4;
      f32x4 v = *(const f32x4*)gp;
      if (k0 + 32 < K)
        __builtin_prefetch(gp + 32, 0, 1);   // global_prefetch_b8 next K-slice
      bf16x4 h4, l4;
#pragma unroll
      for (int j = 0; j < 4; ++j) {
        float  f = v[j];
        __bf16 h = (__bf16)f;
        __bf16 l = (__bf16)(f - (float)h);
        h4[j] = h;
        l4[j] = l;
      }
      *(bf16x4*)&sh_hi[row * LDS_PITCH + f4 * 4] = h4;
      *(bf16x4*)&sh_lo[row * LDS_PITCH + f4 * 4] = l4;
    }
    __syncthreads();

    // ---- A fragments from LDS (ISA 16-bit A 16x32 layout) ------------------
    v16bf ah[4], al[4];
#pragma unroll
    for (int mt = 0; mt < 4; ++mt) {
      const int rowoff = (wave_m * 64 + mt * 16 + laneLow) * LDS_PITCH + laneHi * 8;
      bf16x8 h0 = *(const bf16x8*)&sh_hi[rowoff];
      bf16x8 h1 = *(const bf16x8*)&sh_hi[rowoff + 16];
      ah[mt] = __builtin_shufflevector(h0, h1, 0,1,2,3,4,5,6,7,8,9,10,11,12,13,14,15);
      bf16x8 l0 = *(const bf16x8*)&sh_lo[rowoff];
      bf16x8 l1 = *(const bf16x8*)&sh_lo[rowoff + 16];
      al[mt] = __builtin_shufflevector(l0, l1, 0,1,2,3,4,5,6,7,8,9,10,11,12,13,14,15);
    }

    // ---- B fragments direct from L2-resident scaled weight -----------------
    v16bf bh[2], bl[2];
#pragma unroll
    for (int nt = 0; nt < 2; ++nt) {
      const size_t off =
          (size_t)(nBase + wave_n * 32 + nt * 16 + laneLow) * K + k0 + laneHi * 8;
      bf16x8 h0 = *(const bf16x8*)(whi + off);
      bf16x8 h1 = *(const bf16x8*)(whi + off + 16);
      bh[nt] = __builtin_shufflevector(h0, h1, 0,1,2,3,4,5,6,7,8,9,10,11,12,13,14,15);
      bf16x8 l0 = *(const bf16x8*)(wlo + off);
      bf16x8 l1 = *(const bf16x8*)(wlo + off + 16);
      bl[nt] = __builtin_shufflevector(l0, l1, 0,1,2,3,4,5,6,7,8,9,10,11,12,13,14,15);
    }

    // ---- 3-product hi/lo WMMA (≈fp32 accuracy) -----------------------------
#pragma unroll
    for (int mt = 0; mt < 4; ++mt)
#pragma unroll
      for (int nt = 0; nt < 2; ++nt) {
        acc[mt][nt] = __builtin_amdgcn_wmma_f32_16x16x32_bf16(
            false, ah[mt], false, bh[nt], (short)0, acc[mt][nt], false, false);
        acc[mt][nt] = __builtin_amdgcn_wmma_f32_16x16x32_bf16(
            false, al[mt], false, bh[nt], (short)0, acc[mt][nt], false, false);
        acc[mt][nt] = __builtin_amdgcn_wmma_f32_16x16x32_bf16(
            false, ah[mt], false, bl[nt], (short)0, acc[mt][nt], false, false);
      }
    __syncthreads();
  }

  // ---- epilogue: ISA 32-bit C/D layout (VGPR i -> M = i + 8*laneHi) --------
#pragma unroll
  for (int mt = 0; mt < 4; ++mt) {
    const int mrow = mBase + wave_m * 64 + mt * 16 + laneHi * 8;
#pragma unroll
    for (int nt = 0; nt < 2; ++nt) {
      const int ncol = nBase + wave_n * 32 + nt * 16 + laneLow;
      float* po = out + (size_t)mrow * N + ncol;
#pragma unroll
      for (int i = 0; i < 8; ++i)
        po[(size_t)i * N] = acc[mt][nt][i];
    }
  }
}

// ---------------------------------------------------------------------------
extern "C" void kernel_launch(void* const* d_in, const int* in_sizes, int n_in,
                              void* d_out, int out_size, void* d_ws, size_t ws_size,
                              hipStream_t stream) {
  const float* x   = (const float*)d_in[0];   // [B,S,D_IN]
  const float* W   = (const float*)d_in[1];   // [D_OUT,D_IN]
  const float* A   = (const float*)d_in[2];   // [R,D_IN]
  const float* Bw  = (const float*)d_in[3];   // [D_OUT,R]
  const float* mag = (const float*)d_in[4];   // [1,D_OUT]

  const int D_OUT = in_sizes[4];
  const int D_IN  = in_sizes[1] / D_OUT;
  const int R     = in_sizes[3] / D_OUT;
  const long long M = (long long)in_sizes[0] / D_IN;   // B*S = 32768

  __bf16* whi = (__bf16*)d_ws;                       // [D_OUT, D_IN] bf16 (2 MB)
  __bf16* wlo = whi + (size_t)D_OUT * D_IN;          // [D_OUT, D_IN] bf16 (2 MB)

  dora_prep_kernel<<<D_OUT, 256, 0, stream>>>(W, A, Bw, mag, whi, wlo, D_IN, R);

  dim3 grid((unsigned)(D_OUT / 128), (unsigned)(M / 128));  // N fastest: x reuse in L2
  dora_gemm_kernel<<<grid, 256, 0, stream>>>(x, whi, wlo, (float*)d_out, D_IN, D_OUT);
}
